// INNModel_30202210025972
// MI455X (gfx1250) — compile-verified, matrix operand order checked by reading, not run on hardware
//
#include <hip/hip_runtime.h>
#include <hip/hip_bf16.h>
#include <math.h>

// Problem constants (match reference)
#define SLEN 4096
#define DIM  256
#define HID  512
#define H4   2048   // 4*H
#define H2   1024   // 2*H
#define NE   2000
#define NL   20
#define NB   64     // blocks per direction for the recurrence

typedef __bf16 bf16_t;
typedef bf16_t v16bf __attribute__((ext_vector_type(16)));
typedef float  v8f   __attribute__((ext_vector_type(8)));

static __device__ __forceinline__ unsigned short f32_to_bf16u(float f) {
  unsigned int u = __float_as_uint(f);
  unsigned int r = u + 0x7FFFu + ((u >> 16) & 1u);   // round-to-nearest-even
  return (unsigned short)(r >> 16);
}
static __device__ __forceinline__ bf16_t us_to_bf(unsigned short u) {
  union { unsigned short u; bf16_t b; } c; c.u = u; return c.b;
}
static __device__ __forceinline__ float sigmoidf(float x) {
  return 1.0f / (1.0f + expf(-x));
}

// ---------------------------------------------------------------------------
// Kernel 0: embedding gather -> bf16, weight conversion -> bf16, state init
// ---------------------------------------------------------------------------
__global__ void prep_kernel(const int* __restrict__ sentence,
                            const float* __restrict__ emb,
                            const float* __restrict__ wi_f32_f,
                            const float* __restrict__ wi_f32_b,
                            unsigned short* __restrict__ x_bf,
                            unsigned short* __restrict__ wif_bf,
                            unsigned short* __restrict__ wib_bf,
                            float* __restrict__ h_buf,          // [2][2][512]
                            unsigned int* __restrict__ counters) // [2]
{
  int i = blockIdx.x * blockDim.x + threadIdx.x;      // up to SLEN*DIM
  if (i < SLEN * DIM) {
    int t = i >> 8;            // /DIM
    int d = i & (DIM - 1);
    x_bf[i] = f32_to_bf16u(emb[(size_t)sentence[t] * DIM + d]);
  }
  if (i < H4 * DIM) {
    wif_bf[i] = f32_to_bf16u(wi_f32_f[i]);
    wib_bf[i] = f32_to_bf16u(wi_f32_b[i]);
  }
  if (i < 2 * 2 * HID) h_buf[i] = 0.0f;
  if (i < 2)           counters[i] = 0u;
}

// ---------------------------------------------------------------------------
// Kernel 1: xW[dir] = x @ Wi[dir].T + b[dir]  via v_wmma_f32_16x16x32_bf16
// Register-blocked: each wave owns a 16x64 output strip (4 col tiles), so the
// A fragment is loaded once per K-step and reused across 4 WMMAs.
// ---------------------------------------------------------------------------
__global__ void gemm_xw_kernel(const unsigned short* __restrict__ x_bf,   // [S,D]
                               const unsigned short* __restrict__ wif,    // [4H,D]
                               const unsigned short* __restrict__ wib,    // [4H,D]
                               const float* __restrict__ b_f,
                               const float* __restrict__ b_b,
                               float* __restrict__ xw_f,                  // [S,4H]
                               float* __restrict__ xw_b)
{
  int wave = (blockIdx.x * blockDim.x + threadIdx.x) >> 5;
  int lane = threadIdx.x & 31;
  // tile decomposition: dir(2) x rowTiles(256) x colGroups(32); group = 64 cols
  int cg = wave & 31;   wave >>= 5;
  int rt = wave & 255;  wave >>= 8;
  int dir = wave;

  const unsigned short* W  = dir ? wib : wif;
  const float*          bp = dir ? b_b : b_f;
  float*                out = dir ? xw_b : xw_f;

  int m = lane & 15;          // A: row within tile; B: col within tile
  int g = lane >> 4;          // K half-select per ISA 16-bit fragment layout
  const unsigned short* arow = x_bf + (size_t)(rt * 16 + m) * DIM + g * 8;
  const unsigned short* brow = W    + (size_t)(cg * 64 + m) * DIM + g * 8;

  v8f acc0 = {}, acc1 = {}, acc2 = {}, acc3 = {};
  for (int kk = 0; kk < 8; ++kk) {
    int k0 = kk * 32;
    union { uint4 u; unsigned short s[8]; } alo, ahi;
    alo.u = *(const uint4*)(arow + k0);
    ahi.u = *(const uint4*)(arow + k0 + 16);
    v16bf a;
#pragma unroll
    for (int i = 0; i < 8; ++i) {
      a[i]     = us_to_bf(alo.s[i]);
      a[i + 8] = us_to_bf(ahi.s[i]);
    }
#pragma unroll
    for (int j = 0; j < 4; ++j) {
      union { uint4 u; unsigned short s[8]; } blo, bhi;
      const unsigned short* bj = brow + (size_t)(j * 16) * DIM;
      blo.u = *(const uint4*)(bj + k0);
      bhi.u = *(const uint4*)(bj + k0 + 16);
      v16bf b;
#pragma unroll
      for (int i = 0; i < 8; ++i) {
        b[i]     = us_to_bf(blo.s[i]);
        b[i + 8] = us_to_bf(bhi.s[i]);
      }
      v8f acc = (j == 0) ? acc0 : (j == 1) ? acc1 : (j == 2) ? acc2 : acc3;
      acc = __builtin_amdgcn_wmma_f32_16x16x32_bf16(
          /*neg_a=*/false, a, /*neg_b=*/false, b,
          /*c_mod=*/(short)0, acc, /*reuse_a=*/false, /*reuse_b=*/false);
      if      (j == 0) acc0 = acc;
      else if (j == 1) acc1 = acc;
      else if (j == 2) acc2 = acc;
      else             acc3 = acc;
    }
  }

  // C/D layout: VGPR g2, lane l -> M = g2 + 8*(l>=16), N = l&15
  int mbase = (lane >> 4) * 8;
#pragma unroll
  for (int j = 0; j < 4; ++j) {
    v8f acc = (j == 0) ? acc0 : (j == 1) ? acc1 : (j == 2) ? acc2 : acc3;
    int n = cg * 64 + j * 16 + (lane & 15);
    float bias = bp[n];
#pragma unroll
    for (int g2 = 0; g2 < 8; ++g2) {
      int row = rt * 16 + g2 + mbase;
      out[(size_t)row * H4 + n] = acc[g2] + bias;
    }
  }
}

// ---------------------------------------------------------------------------
// Kernel 2: the bidirectional LSTM recurrence.
// 128 blocks: blocks 0..63 forward, 64..127 backward. Each block owns 8
// hidden dims (32 gate rows). Per step: fp32 GEMV chunk against L2-resident
// Wh, gate nonlinearities, write h to a double-buffered global h exchange,
// then a monotonic atomic-counter barrier across the direction's 64 blocks.
// xW (64 MB, HBM-streamed) is prefetched one step ahead (global_prefetch_b8).
// ---------------------------------------------------------------------------
__global__ void lstm_kernel(const float* __restrict__ xw_f,
                            const float* __restrict__ xw_b,
                            const float* __restrict__ wh_f,
                            const float* __restrict__ wh_b,
                            float* __restrict__ blstm,       // [S, 2H]
                            float* __restrict__ h_buf,       // [2][2][512]
                            unsigned int* __restrict__ counters)
{
  __shared__ float h_s[HID];
  __shared__ float z_s[32];
  __shared__ float c_s[8];

  int dir = blockIdx.x >> 6;
  int b   = blockIdx.x & 63;
  int tid = threadIdx.x;

  const float* xw = dir ? xw_b : xw_f;
  const float* wh = dir ? wh_b : wh_f;
  unsigned int* cnt = counters + dir;
  float* hb = h_buf + dir * (2 * HID);

  int j0   = b * 8;                 // first hidden dim owned by this block
  int rowq = tid >> 3;              // 0..31 : which of the 32 gate rows
  int sub  = tid & 7;               // 0..7  : K-slice within the row
  int gate = rowq >> 3;             // 0..3  (i,f,g,o)
  int jj   = rowq & 7;              // 0..7
  int zrow = gate * HID + j0 + jj;  // global gate-row index in [0,2048)
  const float* whrow = wh + (size_t)zrow * HID;

  if (tid < 8) c_s[tid] = 0.0f;
  __syncthreads();

  for (int t = 0; t < SLEN; ++t) {
    int tt = dir ? (SLEN - 1 - t) : t;
    // hide HBM latency: prefetch next step's xW row while GEMV runs
    if (t + 1 < SLEN) {
      int tn = dir ? (SLEN - 2 - t) : (t + 1);
      __builtin_prefetch(xw + (size_t)tn * H4 + zrow, 0, 3);
    }
    const float* hr = hb + (t & 1) * HID;
    h_s[tid]       = hr[tid];
    h_s[tid + 256] = hr[tid + 256];
    __syncthreads();

    float sum = 0.0f;
#pragma unroll 8
    for (int k = sub; k < HID; k += 8)
      sum = fmaf(whrow[k], h_s[k], sum);
    sum += __shfl_xor(sum, 1);
    sum += __shfl_xor(sum, 2);
    sum += __shfl_xor(sum, 4);
    if (sub == 0) z_s[rowq] = sum + xw[(size_t)tt * H4 + zrow];
    __syncthreads();

    if (tid < 8) {
      float iv = sigmoidf(z_s[tid]);
      float fv = sigmoidf(z_s[8 + tid]);
      float gv = tanhf(z_s[16 + tid]);
      float ov = sigmoidf(z_s[24 + tid]);
      float c  = fv * c_s[tid] + iv * gv;
      c_s[tid] = c;
      float h  = ov * tanhf(c);
      hb[((t + 1) & 1) * HID + j0 + tid] = h;
      blstm[(size_t)tt * H2 + dir * HID + j0 + tid] = h;
      __threadfence();   // release h before signaling
    }
    __syncthreads();

    if (tid == 0) {
      atomicAdd(cnt, 1u);
      unsigned int target = (unsigned int)(NB * (t + 1));
      while (atomicAdd(cnt, 0u) < target)
        __builtin_amdgcn_s_sleep(1);
      __threadfence();   // acquire other blocks' h
    }
    __syncthreads();
  }
}

// ---------------------------------------------------------------------------
// Kernel 3: attention scores, one wave per sequence position
// ---------------------------------------------------------------------------
__global__ void scores_kernel(const float* __restrict__ blstm,
                              const float* __restrict__ attn_w,
                              const float* __restrict__ attn_b,
                              float* __restrict__ scores)
{
  int wave = (blockIdx.x * blockDim.x + threadIdx.x) >> 5;
  int lane = threadIdx.x & 31;
  const float* row = blstm + (size_t)wave * H2;
  float sum = 0.0f;
  for (int k = lane; k < H2; k += 32)
    sum = fmaf(row[k], attn_w[k], sum);
#pragma unroll
  for (int off = 16; off; off >>= 1) sum += __shfl_xor(sum, off);
  if (lane == 0) scores[wave] = sum + attn_b[0];
}

// ---------------------------------------------------------------------------
// Kernel 4: per-entity masked softmax over span scores + weighted row gather
// ---------------------------------------------------------------------------
__global__ void entity_kernel(const int* __restrict__ ent_idx,   // [E,L]
                              const float* __restrict__ scores,
                              const float* __restrict__ blstm,
                              float* __restrict__ out)           // [E,2H]
{
  __shared__ float w_s[NL];
  __shared__ int   id_s[NL];
  int e = blockIdx.x;
  int tid = threadIdx.x;
  if (tid < NL) {
    int ix = ent_idx[e * NL + tid];
    id_s[tid] = ix < 0 ? 0 : ix;
    w_s[tid]  = ix < 0 ? -1e9f : scores[ix < 0 ? 0 : ix];
  }
  __syncthreads();
  if (tid == 0) {
    float m = w_s[0];
    for (int l = 1; l < NL; ++l) m = fmaxf(m, w_s[l]);
    float ssum = 0.0f;
    for (int l = 0; l < NL; ++l) { float ev = expf(w_s[l] - m); w_s[l] = ev; ssum += ev; }
    float inv = 1.0f / ssum;
    for (int l = 0; l < NL; ++l) w_s[l] *= inv;
  }
  __syncthreads();
  for (int d = tid; d < H2; d += 256) {
    float acc = 0.0f;
#pragma unroll
    for (int l = 0; l < NL; ++l)
      acc = fmaf(w_s[l], blstm[(size_t)id_s[l] * H2 + d], acc);
    out[(size_t)e * H2 + d] = acc;
  }
}

// ---------------------------------------------------------------------------
extern "C" void kernel_launch(void* const* d_in, const int* in_sizes, int n_in,
                              void* d_out, int out_size, void* d_ws, size_t ws_size,
                              hipStream_t stream) {
  const int*   sentence = (const int*)d_in[0];
  const int*   ent_idx  = (const int*)d_in[1];
  const float* emb      = (const float*)d_in[2];
  const float* Wi_f     = (const float*)d_in[3];
  const float* Wh_f     = (const float*)d_in[4];
  const float* b_f      = (const float*)d_in[5];
  const float* Wi_b     = (const float*)d_in[6];
  const float* Wh_b     = (const float*)d_in[7];
  const float* b_b      = (const float*)d_in[8];
  const float* attn_w   = (const float*)d_in[9];
  const float* attn_b   = (const float*)d_in[10];
  float*       out      = (float*)d_out;

  char* ws = (char*)d_ws;
  size_t off = 0;
  auto alloc = [&](size_t bytes) -> char* {
    char* p = ws + off;
    off += (bytes + 255) & ~(size_t)255;
    return p;
  };
  unsigned short* x_bf   = (unsigned short*)alloc((size_t)SLEN * DIM * 2);
  unsigned short* wif_bf = (unsigned short*)alloc((size_t)H4 * DIM * 2);
  unsigned short* wib_bf = (unsigned short*)alloc((size_t)H4 * DIM * 2);
  float* xw_f   = (float*)alloc((size_t)SLEN * H4 * 4);
  float* xw_b   = (float*)alloc((size_t)SLEN * H4 * 4);
  float* blstm  = (float*)alloc((size_t)SLEN * H2 * 4);
  float* scores = (float*)alloc((size_t)SLEN * 4);
  float* h_buf  = (float*)alloc((size_t)2 * 2 * HID * 4);
  unsigned int* counters = (unsigned int*)alloc(256);

  // 0: gather embeddings to bf16, convert Wi to bf16, init h/counters
  prep_kernel<<<(SLEN * DIM) / 256, 256, 0, stream>>>(
      sentence, emb, Wi_f, Wi_b, x_bf, wif_bf, wib_bf, h_buf, counters);

  // 1: input projections via WMMA bf16, 16x64 strip per wave
  // waves = 2 dirs * 256 row-tiles * 32 col-groups = 16384 -> 2048 blocks
  gemm_xw_kernel<<<2048, 256, 0, stream>>>(
      x_bf, wif_bf, wib_bf, b_f, b_b, xw_f, xw_b);

  // 2: sequential BiLSTM recurrence, 64 cooperating blocks per direction
  lstm_kernel<<<2 * NB, 256, 0, stream>>>(
      xw_f, xw_b, Wh_f, Wh_b, blstm, h_buf, counters);

  // 3: attention scores (wave per row): 4096 waves = 512 blocks
  scores_kernel<<<(SLEN * 32) / 256, 256, 0, stream>>>(
      blstm, attn_w, attn_b, scores);

  // 4: entity span softmax + aggregation
  entity_kernel<<<NE, 256, 0, stream>>>(ent_idx, scores, blstm, out);
}
